// AttentionBlock_19378892439881
// MI455X (gfx1250) — matrix operand order repaired and checked by reading, not compiled, not run on hardware
//
#include <hip/hip_runtime.h>
#include <hip/hip_bf16.h>

// ---------------------------------------------------------------------------
// AttentionBlock for MI455X (gfx1250): bf16 WMMA pipeline + flash attention.
//   B=2, L=2048, D=1024, H=16, dh=64.  All matmuls on v_wmma_f32_16x16x32_bf16.
//   Non-transposed LDS staging uses GLOBAL_LOAD_ASYNC_TO_LDS_B128 (ASYNCcnt).
// ---------------------------------------------------------------------------

typedef __attribute__((ext_vector_type(16))) __bf16 v16bf;
typedef __attribute__((ext_vector_type(8)))  float  v8f;

union Frag16 {
    v16bf v;
    unsigned int u[8];
};

// fp32 -> bf16 round-to-nearest-even
__device__ __forceinline__ unsigned short f2bf(float f) {
    unsigned int u = __float_as_uint(f);
    u += 0x7FFFu + ((u >> 16) & 1u);
    return (unsigned short)(u >> 16);
}

// A-matrix (16x32, 16-bit) element-pair K index for VGPR v, half hi (lane>=16)
__device__ __forceinline__ int kpairA(int v, int hi) {
    return ((v & 3) << 1) + ((v >> 2) << 4) + (hi << 3);
}
// B-matrix (32x16, 16-bit) element-pair K index
__device__ __forceinline__ int kpairB(int v, int hi) {
    return ((v & 3) << 1) + ((v >> 2) << 4) + (hi << 3);
}

// Async 16-byte global -> LDS copy (per lane), tracked by ASYNCcnt.
// LDS operand: byte offset within the block's LDS allocation == low 32 bits of
// the generic pointer to a __shared__ object (AS3 offset).
__device__ __forceinline__ void async_copy_b128(void* lds_dst, const void* gsrc) {
    unsigned loff = (unsigned)(size_t)lds_dst;
    asm volatile("global_load_async_to_lds_b128 %0, %1, off"
                 :: "v"(loff), "v"(gsrc)
                 : "memory");
}
__device__ __forceinline__ void wait_async0() {
    asm volatile("s_wait_asynccnt 0x0" ::: "memory");
}

// ---------------------------------------------------------------------------
// Kernel 1: elementwise fp32 -> bf16
// ---------------------------------------------------------------------------
__global__ __launch_bounds__(256) void f32_to_bf16_kernel(
    const float* __restrict__ in, unsigned short* __restrict__ out, int n) {
    int i = blockIdx.x * 256 + threadIdx.x;
    if (i < n) out[i] = f2bf(in[i]);
}

// ---------------------------------------------------------------------------
// Kernel 2/4: bf16 GEMM, C[M=4096, N=1024] = A[M,1024] @ W[1024,N] + bias.
//   Block tile 128x64, 256 threads = 8 waves; wave w -> rows [16w,16w+16),
//   4 accumulators across the 64 N columns. K staged in LDS 32 at a time.
//   A tile staged via async global->LDS; W tile manually transposed.
//   mode 0: store fp32 to of32[row*1024+col]
//   mode 1: store bf16*scale into head-major [B=2,H=16,L=2048,dh=64]
// ---------------------------------------------------------------------------
__global__ __launch_bounds__(256) void gemm_bf16_128x64(
    const unsigned short* __restrict__ A, const unsigned short* __restrict__ W,
    const float* __restrict__ bias, float scale, int mode,
    unsigned short* __restrict__ obf, float* __restrict__ of32) {
    __shared__ unsigned short Al[128 * 40];   // A tile 128x32, stride 40 (16B rows)
    __shared__ unsigned short Bt[64 * 34];    // W tile transposed [n][k], stride 34

    const int tid  = threadIdx.x;
    const int lane = tid & 31;
    const int wave = tid >> 5;
    const int hi   = lane >> 4;
    const int l16  = lane & 15;
    const int bm   = blockIdx.x * 128;
    const int bn   = blockIdx.y * 64;

    v8f acc[4] = {};

    for (int kt = 0; kt < 32; ++kt) {
        const int k0 = kt * 32;
        __syncthreads();
        // cooperative A tile: 512 x b128 async copies, 2 per thread
#pragma unroll
        for (int s = 0; s < 2; ++s) {
            int i  = tid + s * 256;
            int r  = i >> 2;
            int c8 = (i & 3) << 3;
            async_copy_b128(&Al[r * 40 + c8],
                            &A[(size_t)(bm + r) * 1024 + k0 + c8]);
        }
        // cooperative W tile load with transpose: [32][64] -> Bt[n][k]
        {
            int r  = tid >> 3;
            int c8 = (tid & 7) << 3;
            int4 t = *(const int4*)&W[(size_t)(k0 + r) * 1024 + bn + c8];
            const unsigned short* p = (const unsigned short*)&t;
#pragma unroll
            for (int j = 0; j < 8; ++j) Bt[(c8 + j) * 34 + r] = p[j];
        }
        wait_async0();
        __syncthreads();

        Frag16 af;
        const int arow = wave * 16 + l16;
#pragma unroll
        for (int v = 0; v < 8; ++v)
            af.u[v] = *(const unsigned int*)&Al[arow * 40 + kpairA(v, hi)];
#pragma unroll
        for (int g = 0; g < 4; ++g) {
            Frag16 bf;
            const int ncol = g * 16 + l16;
#pragma unroll
            for (int v = 0; v < 8; ++v)
                bf.u[v] = *(const unsigned int*)&Bt[ncol * 34 + kpairB(v, hi)];
            acc[g] = __builtin_amdgcn_wmma_f32_16x16x32_bf16(
                false, af.v, false, bf.v, (short)0, acc[g], false, false);
        }
    }

    // store: C layout row = v + 8*hi, col = lane&15 within each 16-col group
#pragma unroll
    for (int g = 0; g < 4; ++g) {
#pragma unroll
        for (int v = 0; v < 8; ++v) {
            const int row = bm + wave * 16 + v + 8 * hi;
            const int col = bn + g * 16 + l16;
            float val = acc[g][v] + bias[col];
            if (mode == 0) {
                of32[(size_t)row * 1024 + col] = val;
            } else {
                const int b = row >> 11, l = row & 2047;
                const int h = col >> 6, d = col & 63;
                obf[(((size_t)(b * 16 + h) * 2048 + l) << 6) + d] = f2bf(val * scale);
            }
        }
    }
}

// ---------------------------------------------------------------------------
// Kernel 3: flash attention. Grid (L/128, B*H); block 256 = 8 waves.
//   Wave owns 16 q-rows; iterates 64 chunks of 32 keys; online softmax fp32.
//   K chunk staged async global->LDS; V chunk manually transposed.
//   O written bf16 to [N=4096, D=1024] (row = b*L+q, col = h*64+d).
// ---------------------------------------------------------------------------
__global__ __launch_bounds__(256) void attn_fwd(
    const unsigned short* __restrict__ Q, const unsigned short* __restrict__ K,
    const unsigned short* __restrict__ V, unsigned short* __restrict__ O) {
    __shared__ unsigned short Kl[32 * 72];       // K chunk [key][d], 16B rows
    __shared__ unsigned short Vt[64 * 34];       // V chunk transposed [d][key]
    __shared__ unsigned short Pl[8 * 16 * 34];   // per-wave P scratch [q][key]

    const int tid  = threadIdx.x;
    const int lane = tid & 31;
    const int wave = tid >> 5;
    const int hi   = lane >> 4;
    const int l16  = lane & 15;
    const int qb   = blockIdx.x * 128;
    const int bh   = blockIdx.y;                 // b*16 + h
    const size_t base_bh = (size_t)bh * 2048 * 64;
    const int pbase = wave * 16 * 34;

    // Q fragments (16 rows x dh=64 -> two K=32 A-fragments), kept in VGPRs
    Frag16 qf[2];
    {
        const int qrow = qb + wave * 16 + l16;
        const unsigned short* qp = Q + base_bh + (size_t)qrow * 64;
#pragma unroll
        for (int ds = 0; ds < 2; ++ds)
#pragma unroll
            for (int v = 0; v < 8; ++v)
                qf[ds].u[v] = *(const unsigned int*)&qp[ds * 32 + kpairA(v, hi)];
    }

    v8f oa[4] = {};
    float mrow[8], lrow[8];
#pragma unroll
    for (int v = 0; v < 8; ++v) { mrow[v] = -3.0e38f; lrow[v] = 0.0f; }

    for (int c = 0; c < 64; ++c) {
        const int kv = c * 32;
        __syncthreads();
        {
            const int r  = tid >> 3;
            const int c8 = (tid & 7) << 3;
            const size_t goff = base_bh + (size_t)(kv + r) * 64 + c8;
            // K chunk: direct async global->LDS (row-major, B-fragment ready)
            async_copy_b128(&Kl[r * 72 + c8], &K[goff]);
            // V chunk: load + transpose into Vt[d][key]
            int4 vx = *(const int4*)&V[goff];
            const unsigned short* p = (const unsigned short*)&vx;
#pragma unroll
            for (int j = 0; j < 8; ++j) Vt[(c8 + j) * 34 + r] = p[j];
            if (c + 1 < 64) {
                __builtin_prefetch(&K[goff + 32 * 64], 0, 1);
                __builtin_prefetch(&V[goff + 32 * 64], 0, 1);
            }
        }
        wait_async0();
        __syncthreads();

        // S = Q @ K^T for 32 keys: two 16x16 C tiles, dh reduced in 2 steps
        v8f s[2];
#pragma unroll
        for (int t = 0; t < 2; ++t) {
            v8f a = {};
#pragma unroll
            for (int ds = 0; ds < 2; ++ds) {
                Frag16 kf;
                const int krow = t * 16 + l16;
#pragma unroll
                for (int v = 0; v < 8; ++v)
                    kf.u[v] = *(const unsigned int*)&Kl[krow * 72 + ds * 32 + kpairB(v, hi)];
                a = __builtin_amdgcn_wmma_f32_16x16x32_bf16(
                    false, qf[ds].v, false, kf.v, (short)0, a, false, false);
            }
            s[t] = a;
        }

        // online softmax across the 32-key chunk (rows in 16-lane halves)
#pragma unroll
        for (int v = 0; v < 8; ++v) {
            float s0 = s[0][v], s1 = s[1][v];
            float mx = fmaxf(s0, s1);
#pragma unroll
            for (int m = 8; m >= 1; m >>= 1) mx = fmaxf(mx, __shfl_xor(mx, m, 16));
            const float mnew  = fmaxf(mrow[v], mx);
            const float alpha = __expf(mrow[v] - mnew);
            const float p0 = __expf(s0 - mnew);
            const float p1 = __expf(s1 - mnew);
            float ps = p0 + p1;
#pragma unroll
            for (int m = 8; m >= 1; m >>= 1) ps += __shfl_xor(ps, m, 16);
            lrow[v] = lrow[v] * alpha + ps;
            mrow[v] = mnew;
            oa[0][v] *= alpha; oa[1][v] *= alpha;
            oa[2][v] *= alpha; oa[3][v] *= alpha;
            const int pr = pbase + (v + 8 * hi) * 34 + l16;
            Pl[pr]      = f2bf(p0);
            Pl[pr + 16] = f2bf(p1);
        }

        // P (16x32) as A-fragment via per-wave LDS re-layout
        Frag16 pf;
#pragma unroll
        for (int v = 0; v < 8; ++v)
            pf.u[v] = *(const unsigned int*)&Pl[pbase + l16 * 34 + kpairA(v, hi)];

        // O += P @ V
#pragma unroll
        for (int g = 0; g < 4; ++g) {
            Frag16 vf;
            const int dcol = g * 16 + l16;
#pragma unroll
            for (int v = 0; v < 8; ++v)
                vf.u[v] = *(const unsigned int*)&Vt[dcol * 34 + kpairB(v, hi)];
            oa[g] = __builtin_amdgcn_wmma_f32_16x16x32_bf16(
                false, pf.v, false, vf.v, (short)0, oa[g], false, false);
        }
    }

    // finalize: O / l, bf16, layout [row = b*L+q][col = h*64+d]
    const int b = bh >> 4, h = bh & 15;
#pragma unroll
    for (int v = 0; v < 8; ++v) {
        const float inv = 1.0f / lrow[v];
        const int row = qb + wave * 16 + v + 8 * hi;
        const size_t rbase = ((size_t)(b * 2048 + row)) * 1024 + h * 64;
#pragma unroll
        for (int g = 0; g < 4; ++g)
            O[rbase + g * 16 + l16] = f2bf(oa[g][v] * inv);
    }
}

// ---------------------------------------------------------------------------
// launch
// ---------------------------------------------------------------------------
extern "C" void kernel_launch(void* const* d_in, const int* in_sizes, int n_in,
                              void* d_out, int out_size, void* d_ws, size_t ws_size,
                              hipStream_t stream) {
    const float* x  = (const float*)d_in[0];
    const float* Wq = (const float*)d_in[1];
    const float* bq = (const float*)d_in[2];
    const float* Wk = (const float*)d_in[3];
    const float* bk = (const float*)d_in[4];
    const float* Wv = (const float*)d_in[5];
    const float* bv = (const float*)d_in[6];
    const float* Wo = (const float*)d_in[7];
    const float* bo = (const float*)d_in[8];
    float* out = (float*)d_out;

    const int N = 4096;           // B*L
    const int D = 1024;
    const int NHD = 2 * 16 * 2048 * 64;   // elements of Q/K/V head-major

    unsigned short* p   = (unsigned short*)d_ws;
    unsigned short* xb  = p;  p += (size_t)N * D;
    unsigned short* Wqb = p;  p += (size_t)D * D;
    unsigned short* Wkb = p;  p += (size_t)D * D;
    unsigned short* Wvb = p;  p += (size_t)D * D;
    unsigned short* Wob = p;  p += (size_t)D * D;
    unsigned short* Qb  = p;  p += (size_t)NHD;
    unsigned short* Kb  = p;  p += (size_t)NHD;
    unsigned short* Vb  = p;  p += (size_t)NHD;
    unsigned short* Ob  = p;  p += (size_t)N * D;

    // 1) convert inputs to bf16
    f32_to_bf16_kernel<<<(N * D + 255) / 256, 256, 0, stream>>>(x,  xb,  N * D);
    f32_to_bf16_kernel<<<(D * D + 255) / 256, 256, 0, stream>>>(Wq, Wqb, D * D);
    f32_to_bf16_kernel<<<(D * D + 255) / 256, 256, 0, stream>>>(Wk, Wkb, D * D);
    f32_to_bf16_kernel<<<(D * D + 255) / 256, 256, 0, stream>>>(Wv, Wvb, D * D);
    f32_to_bf16_kernel<<<(D * D + 255) / 256, 256, 0, stream>>>(Wo, Wob, D * D);

    // 2) Q/K/V projections (Q,K pre-scaled by dh^-0.25 = 64^-0.25)
    const float qkscale = 0.35355339059327373f;
    dim3 gg(32, 16), blk(256);
    gemm_bf16_128x64<<<gg, blk, 0, stream>>>(xb, Wqb, bq, qkscale, 1, Qb, nullptr);
    gemm_bf16_128x64<<<gg, blk, 0, stream>>>(xb, Wkb, bk, qkscale, 1, Kb, nullptr);
    gemm_bf16_128x64<<<gg, blk, 0, stream>>>(xb, Wvb, bv, 1.0f,    1, Vb, nullptr);

    // 3) fused flash attention
    attn_fwd<<<dim3(16, 32), blk, 0, stream>>>(Qb, Kb, Vb, Ob);

    // 4) output projection -> fp32 d_out
    gemm_bf16_128x64<<<gg, blk, 0, stream>>>(Ob, Wob, bo, 1.0f, 0, nullptr, out);
}
